// MultiHeadSelfAttention_66640712564946
// MI455X (gfx1250) — compile-verified, matrix operand order checked by reading, not run on hardware
//
#include <hip/hip_runtime.h>
#include <stdint.h>

// Problem constants
#define Bn 2
#define Ln 2048
#define Dn 1024
#define Hn 16
#define HDn 64
#define BLn (Bn * Ln)

typedef __attribute__((ext_vector_type(16))) __bf16 v16bf;
typedef __attribute__((ext_vector_type(4)))  __bf16 v4bf;
typedef __attribute__((ext_vector_type(8)))  float  v8f;

__device__ __forceinline__ v8f wmma_bf16(v16bf a, v16bf b, v8f c) {
  return __builtin_amdgcn_wmma_f32_16x16x32_bf16(
      /*neg_a=*/false, a, /*neg_b=*/false, b,
      /*c_mod=*/(short)0, c, /*reuse_a=*/false, /*reuse_b=*/false);
}

// Async copy of 16 bytes: global (SGPR base + u32 VGPR offset) -> LDS offset.
// Tracked by ASYNCcnt; no VGPR data round-trip.
__device__ __forceinline__ void async_ld16(uint32_t lds_off, const void* sbase,
                                           uint32_t goff) {
  asm volatile("global_load_async_to_lds_b128 %0, %1, %2"
               :: "v"(lds_off), "v"(goff), "s"(sbase)
               : "memory");
}
__device__ __forceinline__ uint32_t lds_off_of(const void* p) {
  // Flat LDS addresses carry the byte offset in the low 32 bits.
  return (uint32_t)(uintptr_t)p;
}

// ---------------------------------------------------------------------------
// Kernel 1: LayerNorm (eps=1e-6) + f32 -> bf16.  One block per row (D=1024).
// ---------------------------------------------------------------------------
__global__ void __launch_bounds__(256)
ln_bf16_kernel(const float* __restrict__ x, const float* __restrict__ gamma,
               const float* __restrict__ beta, __bf16* __restrict__ out) {
  const int row = blockIdx.x;
  const int tid = threadIdx.x;
  const float4 v = ((const float4*)(x + (size_t)row * Dn))[tid];

  float s  = v.x + v.y + v.z + v.w;
  float ss = v.x * v.x + v.y * v.y + v.z * v.z + v.w * v.w;
#pragma unroll
  for (int m = 1; m < 32; m <<= 1) {
    s  += __shfl_xor(s,  m, 32);
    ss += __shfl_xor(ss, m, 32);
  }
  __shared__ float sred[8], ssred[8];
  const int wid = tid >> 5;
  if ((tid & 31) == 0) { sred[wid] = s; ssred[wid] = ss; }
  __syncthreads();
  s  = sred[tid & 7];
  ss = ssred[tid & 7];
#pragma unroll
  for (int m = 1; m < 8; m <<= 1) {
    s  += __shfl_xor(s,  m, 32);
    ss += __shfl_xor(ss, m, 32);
  }
  const float mu = s * (1.0f / Dn);
  const float var = ss * (1.0f / Dn) - mu * mu;
  const float r = rsqrtf(var + 1e-6f);

  const float4 g = ((const float4*)gamma)[tid];
  const float4 bt = ((const float4*)beta)[tid];
  v4bf y;
  y[0] = (__bf16)((v.x - mu) * r * g.x + bt.x);
  y[1] = (__bf16)((v.y - mu) * r * g.y + bt.y);
  y[2] = (__bf16)((v.z - mu) * r * g.z + bt.z);
  y[3] = (__bf16)((v.w - mu) * r * g.w + bt.w);
  *(v4bf*)(out + (size_t)row * Dn + tid * 4) = y;
}

// ---------------------------------------------------------------------------
// Kernel 2: f32 -> bf16 conversion for weights (vectorized x4)
// ---------------------------------------------------------------------------
__global__ void __launch_bounds__(256)
cvt_bf16_kernel(const float* __restrict__ src, __bf16* __restrict__ dst, int n4) {
  int i = blockIdx.x * blockDim.x + threadIdx.x;
  if (i >= n4) return;
  const float4 v = ((const float4*)src)[i];
  v4bf y;
  y[0] = (__bf16)v.x; y[1] = (__bf16)v.y; y[2] = (__bf16)v.z; y[3] = (__bf16)v.w;
  ((v4bf*)dst)[i] = y;
}

// ---------------------------------------------------------------------------
// Kernel 3/5: WMMA GEMM  Y = X[4096x1024] * W^T[1024x1024]
//   Block = 256 threads = 8 waves (4x2), block tile 128x64, wave tile 32x32.
//   K-tiles (128x32 X, 64x32 W) double-buffered in LDS via async-to-LDS.
//   MODE 0: store bf16 to [B,H,L,64]   (Q / K head layout)
//   MODE 1: store bf16 to [B,H,64,L]   (V transposed head layout)
//   MODE 2: store f32 row-major + residual add (output projection)
// ---------------------------------------------------------------------------
template <int MODE>
__global__ void __launch_bounds__(256)
gemm_wmma_kernel(const __bf16* __restrict__ X, const __bf16* __restrict__ W,
                 void* __restrict__ OUT, const float* __restrict__ RES) {
  __shared__ __align__(16) __bf16 sx[2][128 * 32];  // 2 x 8 KB
  __shared__ __align__(16) __bf16 sw[2][64 * 32];   // 2 x 4 KB

  const int tid  = threadIdx.x;
  const int lane = tid & 31;
  const int wave = tid >> 5;
  const int wm = wave & 3, wn = wave >> 2;
  const int Mblk = blockIdx.y * 128;
  const int Nblk = blockIdx.x * 64;
  const int lm = lane & 15;
  const int hi = lane >> 4;

  // Per-thread async-fill assignments (16B chunks).
  // X tile: 512 chunks (128 rows x 4), thread does c = tid, tid+256.
  // W tile: 256 chunks (64 rows x 4),  thread does c = tid.
  const uint32_t xc0 = tid, xc1 = tid + 256, wc0 = tid;
  const uint32_t xrow0 = xc0 >> 2, xsub0 = (xc0 & 3) * 16;
  const uint32_t xrow1 = xc1 >> 2, xsub1 = (xc1 & 3) * 16;
  const uint32_t wrow0 = wc0 >> 2, wsub0 = (wc0 & 3) * 16;

  const uint32_t sx_base[2] = {lds_off_of(&sx[0][0]), lds_off_of(&sx[1][0])};
  const uint32_t sw_base[2] = {lds_off_of(&sw[0][0]), lds_off_of(&sw[1][0])};

  auto issue_tile = [&](int buf, uint32_t kbyte) {
    async_ld16(sx_base[buf] + xc0 * 16, X, (Mblk + xrow0) * 2048u + kbyte + xsub0);
    async_ld16(sx_base[buf] + xc1 * 16, X, (Mblk + xrow1) * 2048u + kbyte + xsub1);
    async_ld16(sw_base[buf] + wc0 * 16, W, (Nblk + wrow0) * 2048u + kbyte + wsub0);
  };

  // Prologue: fill buffer 0 (k-step 0).
  issue_tile(0, 0);

  v8f acc00 = {}, acc01 = {}, acc10 = {}, acc11 = {};
  const int NSTEP = Dn / 32;  // 32
#pragma unroll 1
  for (int s = 0; s < NSTEP; ++s) {
    const int buf = s & 1;
    __syncthreads();  // close prior readers of buffer (s+1)&1
    if (s + 1 < NSTEP) {
      issue_tile((s + 1) & 1, (uint32_t)(s + 1) * 64u);
      asm volatile("s_wait_asynccnt 0x3" ::: "memory");  // allow the 3 just issued
    } else {
      asm volatile("s_wait_asynccnt 0x0" ::: "memory");
    }
    __syncthreads();  // current buffer resident for all waves

    const __bf16* xa = &sx[buf][0];
    const __bf16* wb = &sw[buf][0];
    const v16bf a0 = *(const v16bf*)(xa + (wm * 32 + lm) * 32 + hi * 16);
    const v16bf a1 = *(const v16bf*)(xa + (wm * 32 + 16 + lm) * 32 + hi * 16);
    const v16bf b0 = *(const v16bf*)(wb + (wn * 32 + lm) * 32 + hi * 16);
    const v16bf b1 = *(const v16bf*)(wb + (wn * 32 + 16 + lm) * 32 + hi * 16);
    acc00 = wmma_bf16(a0, b0, acc00);
    acc01 = wmma_bf16(a0, b1, acc01);
    acc10 = wmma_bf16(a1, b0, acc10);
    acc11 = wmma_bf16(a1, b1, acc11);
  }

  const int M0 = Mblk + wm * 32;
  const int N0 = Nblk + wn * 32;
  v8f accs[2][2] = {{acc00, acc01}, {acc10, acc11}};
#pragma unroll
  for (int i = 0; i < 2; i++)
#pragma unroll
    for (int j = 0; j < 2; j++)
#pragma unroll
      for (int v = 0; v < 8; v++) {
        const int row = M0 + i * 16 + v + 8 * hi;
        const int col = N0 + j * 16 + lm;
        const float val = accs[i][j][v];
        if (MODE == 2) {
          ((float*)OUT)[(size_t)row * Dn + col] =
              val + RES[(size_t)row * Dn + col];
        } else {
          const int b = row >> 11, l = row & (Ln - 1);
          const int h = col >> 6,  dd = col & (HDn - 1);
          size_t idx;
          if (MODE == 0)
            idx = ((size_t)((b * Hn + h) * Ln + l)) * HDn + dd;
          else
            idx = ((size_t)((b * Hn + h) * HDn + dd)) * Ln + l;
          ((__bf16*)OUT)[idx] = (__bf16)val;
        }
      }
}

// ---------------------------------------------------------------------------
// Kernel 4: Flash attention, one (b,h,q-tile-of-64) per block, 4 waves.
//   K/V 64-wide tiles double-buffered in LDS via async-to-LDS (shared by all
//   4 waves).  Online softmax; P staged through per-wave LDS region.
// ---------------------------------------------------------------------------
__global__ void __launch_bounds__(128)
flash_attn_kernel(const __bf16* __restrict__ Q, const __bf16* __restrict__ K,
                  const __bf16* __restrict__ Vt, const uint8_t* __restrict__ mask,
                  __bf16* __restrict__ out) {
  __shared__ __align__(16) __bf16 sk[2][64 * 64];   // 2 x 8 KB  [key][d]
  __shared__ __align__(16) __bf16 sv[2][64 * 64];   // 2 x 8 KB  [d][k]
  __shared__ __align__(32) __bf16 lds_p[4][16 * 64];  // 8 KB

  const int tid  = threadIdx.x;
  const int lane = tid & 31;
  const int wave = tid >> 5;
  const int qt = blockIdx.x;  // 0..31
  const int h  = blockIdx.y;  // 0..15
  const int b  = blockIdx.z;  // 0..1
  const int q0 = qt * 64 + wave * 16;
  const int lm = lane & 15;
  const int hi = lane >> 4;

  const __bf16* Kh = K  + (size_t)(b * Hn + h) * Ln * HDn;  // [L][64]
  const __bf16* Vh = Vt + (size_t)(b * Hn + h) * HDn * Ln;  // [64][L]
  const uint8_t* mrow = mask + (size_t)b * Ln * Ln;

  // Per-thread async-fill assignments: 512 16B chunks each for K and V,
  // 128 threads -> 4 chunks per tensor per thread.
  const uint32_t sk_base[2] = {lds_off_of(&sk[0][0]), lds_off_of(&sk[1][0])};
  const uint32_t sv_base[2] = {lds_off_of(&sv[0][0]), lds_off_of(&sv[1][0])};

  auto issue_tile = [&](int buf, int kt) {
#pragma unroll
    for (int i = 0; i < 4; i++) {
      const uint32_t c = tid + i * 128;               // chunk id 0..511
      const uint32_t r = c >> 3, sub = (c & 7) * 16;  // row, 16B sub-chunk
      // K tile: rows are keys kt+r, 128B each, contiguous in d.
      async_ld16(sk_base[buf] + c * 16, Kh, (kt + r) * 128u + sub);
      // V tile: rows are d = r of Vt (stride L*2 bytes), window at kt.
      async_ld16(sv_base[buf] + c * 16, Vh, r * (Ln * 2u) + kt * 2u + sub);
    }
  };

  // Q A-fragments (16 rows x 64 K), resident for the whole loop
  const __bf16* qp = Q + (size_t)(b * Hn + h) * Ln * HDn +
                     (size_t)(q0 + lm) * HDn + hi * 16;
  const v16bf qa0 = *(const v16bf*)(qp);
  const v16bf qa1 = *(const v16bf*)(qp + 32);

  float rm[8], rl[8];
#pragma unroll
  for (int v = 0; v < 8; v++) { rm[v] = -3.0e38f; rl[v] = 0.0f; }
  v8f o[4] = {};
  __bf16* pb = lds_p[wave];

  issue_tile(0, 0);  // prologue

  const int NT = Ln / 64;  // 32 tiles
#pragma unroll 1
  for (int t = 0; t < NT; ++t) {
    const int kt = t * 64;
    const int buf = t & 1;
    __syncthreads();  // close prior readers of buffer (t+1)&1
    if (t + 1 < NT) {
      issue_tile((t + 1) & 1, kt + 64);
      asm volatile("s_wait_asynccnt 0x8" ::: "memory");  // allow the 8 just issued
    } else {
      asm volatile("s_wait_asynccnt 0x0" ::: "memory");
    }
    __syncthreads();  // current K/V tile resident for all waves

    // ---- S = Q K^T for this 16x64 tile (B-frags from LDS) ----
    const __bf16* kb = &sk[buf][0];
    v8f s[4] = {};
#pragma unroll
    for (int nt = 0; nt < 4; nt++) {
      const __bf16* kp = kb + (nt * 16 + lm) * 64 + hi * 16;
      s[nt] = wmma_bf16(qa0, *(const v16bf*)(kp), s[nt]);
      s[nt] = wmma_bf16(qa1, *(const v16bf*)(kp + 32), s[nt]);
    }

    // ---- scale (1/sqrt(D)=1/32), mask, tile row-max ----
    float tm[8];
#pragma unroll
    for (int v = 0; v < 8; v++) tm[v] = -3.0e38f;
#pragma unroll
    for (int nt = 0; nt < 4; nt++) {
      const int kc = kt + nt * 16 + lm;
#pragma unroll
      for (int v = 0; v < 8; v++) {
        const int qr = q0 + v + 8 * hi;
        float sv_ = s[nt][v] * (1.0f / 32.0f);
        const uint8_t mk = mrow[(size_t)qr * Ln + kc];
        sv_ = mk ? sv_ : -1.0e9f;
        s[nt][v] = sv_;
        tm[v] = fmaxf(tm[v], sv_);
      }
    }
#pragma unroll
    for (int v = 0; v < 8; v++)
#pragma unroll
      for (int m = 1; m < 16; m <<= 1)
        tm[v] = fmaxf(tm[v], __shfl_xor(tm[v], m, 32));

    // ---- online softmax update ----
    float scale[8], ts[8];
#pragma unroll
    for (int v = 0; v < 8; v++) {
      const float mn = fmaxf(rm[v], tm[v]);
      scale[v] = __expf(rm[v] - mn);
      rm[v] = mn;
      ts[v] = 0.0f;
    }
#pragma unroll
    for (int nt = 0; nt < 4; nt++)
#pragma unroll
      for (int v = 0; v < 8; v++) {
        const float p = __expf(s[nt][v] - rm[v]);
        s[nt][v] = p;
        ts[v] += p;
      }
#pragma unroll
    for (int v = 0; v < 8; v++)
#pragma unroll
      for (int m = 1; m < 16; m <<= 1)
        ts[v] += __shfl_xor(ts[v], m, 32);
#pragma unroll
    for (int v = 0; v < 8; v++) rl[v] = rl[v] * scale[v] + ts[v];
#pragma unroll
    for (int dt = 0; dt < 4; dt++)
#pragma unroll
      for (int v = 0; v < 8; v++) o[dt][v] *= scale[v];

    // ---- P (C-layout f32) -> LDS bf16 (row-major 16x64, per-wave region) ----
#pragma unroll
    for (int nt = 0; nt < 4; nt++)
#pragma unroll
      for (int v = 0; v < 8; v++)
        pb[(v + 8 * hi) * 64 + nt * 16 + lm] = (__bf16)s[nt][v];
    // same-wave DS ops are in-order; wait + compiler barrier before re-reading
    asm volatile("s_wait_dscnt 0" ::: "memory");

    // ---- O += P V (A-frag from P LDS, B-frag from staged V tile) ----
    const __bf16* vb = &sv[buf][0];
#pragma unroll
    for (int ks = 0; ks < 2; ks++) {
      const v16bf pa = *(const v16bf*)(pb + lm * 64 + ks * 32 + hi * 16);
#pragma unroll
      for (int dt = 0; dt < 4; dt++) {
        const __bf16* vp = vb + (dt * 16 + lm) * 64 + ks * 32 + hi * 16;
        o[dt] = wmma_bf16(pa, *(const v16bf*)vp, o[dt]);
      }
    }
    asm volatile("s_wait_dscnt 0" ::: "memory");  // WAR guard before next P write
  }

  // ---- finalize: O / l, write to [B*L, D] bf16 (head h occupies cols h*64..) ----
#pragma unroll
  for (int dt = 0; dt < 4; dt++)
#pragma unroll
    for (int v = 0; v < 8; v++) {
      const int qr = q0 + v + 8 * hi;
      const int col = h * HDn + dt * 16 + lm;
      out[(size_t)(b * Ln + qr) * Dn + col] = (__bf16)(o[dt][v] / rl[v]);
    }
}

// ---------------------------------------------------------------------------
// Host-side launcher
// ---------------------------------------------------------------------------
extern "C" void kernel_launch(void* const* d_in, const int* in_sizes, int n_in,
                              void* d_out, int out_size, void* d_ws, size_t ws_size,
                              hipStream_t stream) {
  const float*   q_in   = (const float*)d_in[0];
  const uint8_t* mask   = (const uint8_t*)d_in[1];
  const float*   w_qs   = (const float*)d_in[2];
  const float*   w_ks   = (const float*)d_in[3];
  const float*   w_vs   = (const float*)d_in[4];
  const float*   w_fc   = (const float*)d_in[5];
  const float*   gamma  = (const float*)d_in[6];
  const float*   beta   = (const float*)d_in[7];
  float*         outp   = (float*)d_out;

  // Workspace layout (bf16 intermediates; ~48 MB total)
  char* ws = (char*)d_ws;
  const size_t MB = 1024ull * 1024ull;
  __bf16* x_ln  = (__bf16*)(ws + 0 * MB);        // [4096,1024]   8 MB
  __bf16* wq_bf = (__bf16*)(ws + 8 * MB);        // [1024,1024]   2 MB
  __bf16* wk_bf = (__bf16*)(ws + 10 * MB);
  __bf16* wv_bf = (__bf16*)(ws + 12 * MB);
  __bf16* wf_bf = (__bf16*)(ws + 14 * MB);
  __bf16* Qh    = (__bf16*)(ws + 16 * MB);       // [B,H,L,64]    8 MB
  __bf16* Kh    = (__bf16*)(ws + 24 * MB);       // [B,H,L,64]    8 MB
  __bf16* Vt    = (__bf16*)(ws + 32 * MB);       // [B,H,64,L]    8 MB
  __bf16* attnO = (__bf16*)(ws + 40 * MB);       // [4096,1024]   8 MB

  // 1) LayerNorm -> bf16
  ln_bf16_kernel<<<BLn, 256, 0, stream>>>(q_in, gamma, beta, x_ln);

  // 2) Weights -> bf16
  const int n4 = (Dn * Dn) / 4;
  const int cblk = (n4 + 255) / 256;
  cvt_bf16_kernel<<<cblk, 256, 0, stream>>>(w_qs, wq_bf, n4);
  cvt_bf16_kernel<<<cblk, 256, 0, stream>>>(w_ks, wk_bf, n4);
  cvt_bf16_kernel<<<cblk, 256, 0, stream>>>(w_vs, wv_bf, n4);
  cvt_bf16_kernel<<<cblk, 256, 0, stream>>>(w_fc, wf_bf, n4);

  // 3) QKV projections (WMMA GEMMs, async-LDS staged)
  dim3 ggrid(Dn / 64, BLn / 128);  // (16, 32)
  gemm_wmma_kernel<0><<<ggrid, 256, 0, stream>>>(x_ln, wq_bf, (void*)Qh, nullptr);
  gemm_wmma_kernel<0><<<ggrid, 256, 0, stream>>>(x_ln, wk_bf, (void*)Kh, nullptr);
  gemm_wmma_kernel<1><<<ggrid, 256, 0, stream>>>(x_ln, wv_bf, (void*)Vt, nullptr);

  // 4) Flash attention
  dim3 agrid(Ln / 64, Hn, Bn);  // (32, 16, 2)
  flash_attn_kernel<<<agrid, 128, 0, stream>>>(Qh, Kh, Vt, mask, attnO);

  // 5) Output projection + residual (f32 out)
  gemm_wmma_kernel<2><<<ggrid, 256, 0, stream>>>(attnO, wf_bf, (void*)outp, q_in);
}
